// VectorQuantizer_44100724195951
// MI455X (gfx1250) — compile-verified, matrix operand order checked by reading, not run on hardware
//
#include <hip/hip_runtime.h>
#include <hip/hip_bf16.h>

// CDNA5 (gfx1250) wave32 WMMA types
typedef __attribute__((ext_vector_type(16))) __bf16 v16bf;
typedef __attribute__((ext_vector_type(8)))  __bf16 v8bf;
typedef __attribute__((ext_vector_type(8)))  float  v8f;
typedef __attribute__((ext_vector_type(4)))  int    v4i;

#define VQ_D    256
#define VQ_K    1024
#define VQ_N    65536           // 16 * 4096 rows
#define VQ_COMMIT_SCALE (1.25f / 16777216.0f)   // (1 + 0.25) / (N*D)

#define VQ_CHUNK_CODES 64
#define VQ_CHUNK_ELEMS (VQ_CHUNK_CODES * VQ_D)   // 16384 bf16 = 32 KB
#define VQ_NCHUNK      (VQ_K / VQ_CHUNK_CODES)   // 16
#define VQ_TILES_PER_CHUNK (VQ_CHUNK_CODES / 16) // 4

// ---- async global->LDS path (gfx1250 ASYNCcnt), with safe fallback --------
#if defined(__has_builtin)
# if __has_builtin(__builtin_amdgcn_global_load_async_to_lds_b128)
#  define VQ_ASYNC 1
# endif
#endif
#ifndef VQ_ASYNC
# define VQ_ASYNC 0
#endif

typedef __attribute__((address_space(1))) v4i* vq_gptr;
typedef __attribute__((address_space(3))) v4i* vq_lptr;

__device__ __forceinline__ void vq_wait_async() {
#if VQ_ASYNC
# if __has_builtin(__builtin_amdgcn_s_wait_asynccnt)
    __builtin_amdgcn_s_wait_asynccnt(0);
# else
    asm volatile("s_wait_asynccnt 0" ::: "memory");
# endif
#endif
}

// Copy one 64-code chunk (32 KB) of the bf16 codebook into LDS.
// 256 threads x 8 sweeps x 16 B.
__device__ __forceinline__ void vq_stage(const __bf16* __restrict__ Etg,
                                         __bf16* dst, int tid) {
#if VQ_ASYNC
    #pragma unroll
    for (int s = 0; s < 8; ++s) {
        const int off = s * 2048 + tid * 8;   // bf16 elements
        v4i* gp = (v4i*)(Etg + off);          // drop const, reinterpret
        v4i* lp = (v4i*)(dst + off);
        __builtin_amdgcn_global_load_async_to_lds_b128(
            (vq_gptr)gp, (vq_lptr)lp, /*offset=*/0, /*cpol=*/0);
    }
#else
    #pragma unroll
    for (int s = 0; s < 8; ++s) {
        const int off = s * 2048 + tid * 8;
        *(uint4*)(dst + off) = *(const uint4*)(Etg + off);
    }
#endif
}

// Load the 4 B fragments of one k-step from the LDS-resident chunk.
__device__ __forceinline__ void vq_load_b(const __bf16* brow, int o,
                                          v16bf b[VQ_TILES_PER_CHUNK]) {
    #pragma unroll
    for (int ntl = 0; ntl < VQ_TILES_PER_CHUNK; ++ntl) {
        const __bf16* ecol = brow + ntl * (16 * VQ_D) + o;
        v8bf lo = *(const v8bf*)(ecol);
        v8bf hi = *(const v8bf*)(ecol + 16);
        #pragma unroll
        for (int i = 0; i < 8; ++i) { b[ntl][i] = lo[i]; b[ntl][8 + i] = hi[i]; }
    }
}

// ---------------------------------------------------------------------------
// Zero the one-hot encodings region (256 MB) and the loss slot each call.
// ---------------------------------------------------------------------------
__global__ __launch_bounds__(256) void vq_init(float4* __restrict__ enc4,
                                               size_t n4,
                                               float* __restrict__ loss) {
    size_t i      = (size_t)blockIdx.x * blockDim.x + threadIdx.x;
    size_t stride = (size_t)gridDim.x * blockDim.x;
    float4 z; z.x = 0.f; z.y = 0.f; z.z = 0.f; z.w = 0.f;
    for (; i < n4; i += stride) enc4[i] = z;
    if (blockIdx.x == 0 && threadIdx.x == 0) *loss = 0.f;
}

// ---------------------------------------------------------------------------
// Prep: Et[k][d]  = bf16(E[d][k])   (row-per-code, WMMA-B friendly)
//       Etf[k][d] = E[d][k]         (fp32 transposed, for coalesced gather)
//       enorm[k]  = sum_d E[d][k]^2
// ---------------------------------------------------------------------------
__global__ __launch_bounds__(256) void vq_prep(const float* __restrict__ E,
                                               __bf16* __restrict__ Et,
                                               float*  __restrict__ Etf,
                                               float*  __restrict__ enorm) {
    int k = blockIdx.x * blockDim.x + threadIdx.x;
    if (k >= VQ_K) return;
    float s = 0.f;
    for (int d = 0; d < VQ_D; ++d) {
        float v = E[(size_t)d * VQ_K + k];
        s += v * v;
        Et [(size_t)k * VQ_D + d] = (__bf16)v;
        Etf[(size_t)k * VQ_D + d] = v;
    }
    enorm[k] = s;
}

// ---------------------------------------------------------------------------
// Main fused kernel. 256 threads = 8 waves; each wave owns 16 rows of X.
//   - codebook streamed through LDS (double-buffered 32 KB chunks, async DMA)
//   - per chunk: ks-outer with 4 persistent accumulators AND a one-step
//     software pipeline on the B fragments: loads for ks+1 are issued before
//     the 4 WMMAs of ks consume the current set (full WMMA-group of latency
//     cover; forces two live staging sets instead of one recycled pair)
//   - argmin over (||e||^2 - 2*sim): ||x||^2 is row-constant -> dropped
//   - strict '<' in hot path (candidate indices strictly increase per slot)
// ---------------------------------------------------------------------------
__global__ __launch_bounds__(256) void vq_main(const float*  __restrict__ X,
                                               const float*  __restrict__ Etf,
                                               const __bf16* __restrict__ Et,
                                               const float*  __restrict__ enorm,
                                               float* __restrict__ out_q,
                                               float* __restrict__ out_enc,
                                               float* __restrict__ out_idx,
                                               float* __restrict__ out_loss) {
    __shared__ __align__(16) __bf16 s_et[2][VQ_CHUNK_ELEMS];  // 2 x 32 KB
    __shared__ int s_idx[8][16];

    const int tid  = threadIdx.x;
    const int lane = tid & 31;
    const int wave = tid >> 5;
    const int g    = lane >> 4;     // half-wave select (K-slice / M-offset)
    const int ln   = lane & 15;     // A row / C column within tile
    const int rowBase = (blockIdx.x * 8 + wave) * 16;

    // ---- A fragments: 16 rows x 256 (bf16), ISA 16-bit A layout:
    //   lane ln holds row M=ln; half g covers K-subgroups g*8..g*8+7 within
    //   each 32-wide K step (elems 0..7 at +0, elems 8..15 at +16).
    v16bf a[8];
    {
        const float* xrow = X + (size_t)(rowBase + ln) * VQ_D;
        #pragma unroll
        for (int ks = 0; ks < 8; ++ks) {
            const int d0 = ks * 32 + g * 8;
            const int d1 = ks * 32 + 16 + g * 8;
            float4 f0 = *(const float4*)(xrow + d0);
            float4 f1 = *(const float4*)(xrow + d0 + 4);
            float4 f2 = *(const float4*)(xrow + d1);
            float4 f3 = *(const float4*)(xrow + d1 + 4);
            float t[16] = { f0.x, f0.y, f0.z, f0.w,  f1.x, f1.y, f1.z, f1.w,
                            f2.x, f2.y, f2.z, f2.w,  f3.x, f3.y, f3.z, f3.w };
            #pragma unroll
            for (int i = 0; i < 16; ++i) a[ks][i] = (__bf16)t[i];
        }
    }

    float bestd[8];
    int   besti[8];
    #pragma unroll
    for (int r = 0; r < 8; ++r) { bestd[r] = 3.4e38f; besti[r] = 0x7fffffff; }

    // prologue: chunk 0 into buffer 0
    vq_stage(Et, &s_et[0][0], tid);
    vq_wait_async();
    __syncthreads();

    for (int c = 0; c < VQ_NCHUNK; ++c) {
        const __bf16* buf = &s_et[c & 1][0];
        if (c + 1 < VQ_NCHUNK)   // prefetch next chunk while computing
            vq_stage(Et + (size_t)(c + 1) * VQ_CHUNK_ELEMS,
                     &s_et[(c + 1) & 1][0], tid);

        v8f acc0 = {}, acc1 = {}, acc2 = {}, acc3 = {};
        const __bf16* brow = buf + ln * VQ_D;        // this lane's B column

        v16bf b[VQ_TILES_PER_CHUNK], bn[VQ_TILES_PER_CHUNK];
        vq_load_b(brow, g * 8, b);                   // preload k-step 0
        #pragma unroll
        for (int ks = 0; ks < 8; ++ks) {
            if (ks < 7)                              // prefetch k-step ks+1
                vq_load_b(brow, (ks + 1) * 32 + g * 8, bn);
            acc0 = __builtin_amdgcn_wmma_f32_16x16x32_bf16(
                false, a[ks], false, b[0], (short)0, acc0, false, false);
            acc1 = __builtin_amdgcn_wmma_f32_16x16x32_bf16(
                false, a[ks], false, b[1], (short)0, acc1, false, false);
            acc2 = __builtin_amdgcn_wmma_f32_16x16x32_bf16(
                false, a[ks], false, b[2], (short)0, acc2, false, false);
            acc3 = __builtin_amdgcn_wmma_f32_16x16x32_bf16(
                false, a[ks], false, b[3], (short)0, acc3, false, false);
            #pragma unroll
            for (int ntl = 0; ntl < VQ_TILES_PER_CHUNK; ++ntl)
                b[ntl] = bn[ntl];                    // renamed away by unroll
        }

        // fold the 4 tiles into the running argmin (outside the WMMA chain)
        #pragma unroll
        for (int ntl = 0; ntl < VQ_TILES_PER_CHUNK; ++ntl) {
            const v8f acc = (ntl == 0) ? acc0 : (ntl == 1) ? acc1
                          : (ntl == 2) ? acc2 : acc3;
            const int   ci = c * VQ_CHUNK_CODES + ntl * 16 + ln;
            const float en = enorm[ci];
            #pragma unroll
            for (int r = 0; r < 8; ++r) {
                // C layout: this lane's slot r is (M = g*8+r, N = ln)
                float score = en - 2.0f * acc[r];   // row term is constant
                if (score < bestd[r]) { bestd[r] = score; besti[r] = ci; }
            }
        }
        vq_wait_async();
        __syncthreads();
    }

    // ---- argmin across the 16 columns each half-wave saw (ties: low idx) --
    #pragma unroll
    for (int r = 0; r < 8; ++r) {
        float bd = bestd[r];
        int   bi = besti[r];
        #pragma unroll
        for (int off = 1; off < 16; off <<= 1) {     // stays within half-wave
            float od = __shfl_xor(bd, off, 32);
            int   oi = __shfl_xor(bi, off, 32);
            if (od < bd || (od == bd && oi < bi)) { bd = od; bi = oi; }
        }
        if (ln == 0) s_idx[wave][g * 8 + r] = bi;    // rows g*8+r
    }
    __syncthreads();

    // ---- outputs: coalesced fp32 gather from transposed codebook ----------
    float lsum = 0.f;
    for (int m = 0; m < 16; ++m) {
        const int row = rowBase + m;
        const int idx = s_idx[wave][m];
        const int d0  = lane * 8;                    // 32 lanes x 8 = 256
        const float* qrow = Etf + (size_t)idx * VQ_D;
        const float* xrow = X + (size_t)row * VQ_D;
        float4 q0 = *(const float4*)(qrow + d0);
        float4 q1 = *(const float4*)(qrow + d0 + 4);
        float4 x0 = *(const float4*)(xrow + d0);
        float4 x1 = *(const float4*)(xrow + d0 + 4);
        *(float4*)(out_q + (size_t)row * VQ_D + d0)     = q0;   // ST fwd
        *(float4*)(out_q + (size_t)row * VQ_D + d0 + 4) = q1;
        float dv;
        dv = q0.x - x0.x; lsum += dv * dv;
        dv = q0.y - x0.y; lsum += dv * dv;
        dv = q0.z - x0.z; lsum += dv * dv;
        dv = q0.w - x0.w; lsum += dv * dv;
        dv = q1.x - x1.x; lsum += dv * dv;
        dv = q1.y - x1.y; lsum += dv * dv;
        dv = q1.z - x1.z; lsum += dv * dv;
        dv = q1.w - x1.w; lsum += dv * dv;
        if (lane == 0) {
            out_enc[(size_t)row * VQ_K + idx] = 1.0f;
            out_idx[row] = (float)idx;
        }
    }
    #pragma unroll
    for (int off = 1; off < 32; off <<= 1) lsum += __shfl_xor(lsum, off, 32);
    if (lane == 0) atomicAdd(out_loss, lsum * VQ_COMMIT_SCALE);
}

// ---------------------------------------------------------------------------
extern "C" void kernel_launch(void* const* d_in, const int* in_sizes, int n_in,
                              void* d_out, int out_size, void* d_ws, size_t ws_size,
                              hipStream_t stream) {
    const float* X = (const float*)d_in[0];   // [16,4096,256] f32
    const float* E = (const float*)d_in[1];   // [256,1024] f32

    float* out      = (float*)d_out;
    float* out_q    = out;                                         // 16,777,216
    float* out_enc  = out + (size_t)16777216;                      // 67,108,864
    float* out_idx  = out + (size_t)16777216 + (size_t)67108864;   // 65,536
    float* out_loss = out_idx + (size_t)65536;                     // 1

    // workspace: | Et bf16 512KB | enorm 4KB | Etf fp32 1MB |
    __bf16* Et    = (__bf16*)d_ws;
    float*  enorm = (float*)((char*)d_ws + (size_t)VQ_K * VQ_D * 2);
    float*  Etf   = (float*)((char*)d_ws + (size_t)VQ_K * VQ_D * 2
                                         + (size_t)VQ_K * 4);

    // 1) zero one-hot region + loss (deterministic across graph replays)
    vq_init<<<4096, 256, 0, stream>>>((float4*)out_enc,
                                      (size_t)VQ_N * VQ_K / 4, out_loss);
    // 2) bf16 codebook transpose + fp32 transpose + code norms
    vq_prep<<<VQ_K / 256, 256, 0, stream>>>(E, Et, Etf, enorm);
    // 3) fused GEMM(argmin) + quantize + one-hot + loss
    //    512 blocks x 8 waves x 16 rows = 65536 rows
    vq_main<<<VQ_N / 128, 256, 0, stream>>>(X, Etf, Et, enorm,
                                            out_q, out_enc, out_idx, out_loss);
}
